// DULLI2_21869973471276
// MI455X (gfx1250) — matrix operand order repaired and checked by reading, not compiled, not run on hardware
//
#include <hip/hip_runtime.h>
#include <hip/hip_bf16.h>
#include <stdint.h>

// ---------------------------------------------------------------------------
// MI455X (gfx1250): VQ -> conv3x3(reflect)+LeakyReLU -> conv1x1 -> VQ
// GEMMs: v_wmma_f32_16x16x32_bf16 (bf16 operands, f32 accumulate).
// A-tile staging for the chunked GEMMs uses the Tensor Data Mover
// (tensor_load_to_lds + s_wait_tensorcnt), double-buffered in LDS.
// softmax(-dist/T) drops the per-row ||f||^2 term: logit = 200*(f.c) - 100*||c||^2
// ---------------------------------------------------------------------------

typedef __attribute__((ext_vector_type(16))) __bf16          v16bf;
typedef __attribute__((ext_vector_type(8)))  float           v8f;
typedef __attribute__((ext_vector_type(8)))  unsigned short  u16x8;
typedef __attribute__((ext_vector_type(4)))  unsigned int    u32x4;
typedef __attribute__((ext_vector_type(8)))  int             i32x8;
typedef __attribute__((ext_vector_type(4)))  int             i32x4;

union V16 { v16bf v; u16x8 h[2]; };

__device__ __forceinline__ unsigned short f2bf(float f) {
  unsigned int u = __builtin_bit_cast(unsigned int, f);
  u += 0x7FFFu + ((u >> 16) & 1u);            // round-to-nearest-even
  return (unsigned short)(u >> 16);
}

__device__ __forceinline__ v8f wmma_bf16(v16bf a, v16bf b, v8f c) {
  // (neg_a, A, neg_b, B, c_mod, C, reuse_a, reuse_b)
  return __builtin_amdgcn_wmma_f32_16x16x32_bf16(false, a, false, b, (short)0, c,
                                                 false, false);
}

// A fragment (16x32 bf16) from row-major source, leading dim `ld` (elements).
// ISA A-layout: lane<16 holds K in {k0..k0+7, k0+16..k0+23}; lane>=16 the +8 runs.
__device__ __forceinline__ v16bf a_frag(const unsigned short* A, int ld, int k0,
                                        int lane) {
  int m  = lane & 15;
  int kh = (lane >> 4) << 3;                  // 0 or 8
  const unsigned short* p = A + (size_t)m * ld + k0 + kh;
  V16 r;
  r.h[0] = *(const u16x8*)(p);
  r.h[1] = *(const u16x8*)(p + 16);
  return r.v;
}

// B fragment from pre-swizzled storage: each 32x16 (KxN) tile = 512 u16,
// lane-major so a lane's 16 values (K = 8*(lane/16) + (e&7) + 16*(e>>3), N = lane%16) are contiguous.
__device__ __forceinline__ v16bf b_frag(const unsigned short* Bp, size_t tile,
                                        int lane) {
  const unsigned short* p = Bp + tile * 512 + lane * 16;
  V16 r;
  r.h[0] = *(const u16x8*)(p);
  r.h[1] = *(const u16x8*)(p + 8);
  return r.v;
}

__device__ __forceinline__ int refl(int i) {            // jnp 'reflect' (no edge dup)
  return i < 0 ? -i : (i > 39 ? 78 - i : i);
}

// --------------------- Tensor Data Mover tile load -------------------------
// 2-D bf16 tile (tile0 elems x tile1 rows, row stride ld elems) -> LDS.
// D# group0: count=1, lds_addr, 57-bit global addr, type=2.
// D# group1: data_size=1 (2B), tensor_dim0=ld, tensor_dim1=2^20,
//            tile_dim0/1, tensor_dim0_stride=ld.
__device__ __forceinline__ void tdm_load_tile_bf16(unsigned lds_off,
                                                   const unsigned short* g,
                                                   unsigned tile0, unsigned tile1,
                                                   unsigned ld) {
  unsigned long long ga = (unsigned long long)(uintptr_t)g;
  u32x4 g0 = { 1u,                                      // count=1 (user descriptor)
               lds_off,                                 // LDS byte address
               (unsigned)ga,                            // global_addr[31:0]
               (unsigned)((ga >> 32) & 0x01FFFFFFu) | 0x80000000u }; // [56:32]|type=2
  i32x8 g1 = { (int)0x00010000u,                        // mask=0, data_size=1 (2B)
               (int)((ld & 0xFFFFu) << 16),             // tensor_dim0[15:0]
               (int)((ld >> 16) & 0xFFFFu),             // tensor_dim0[31:16], td1 lo=0
               (int)(16u | (tile0 << 16)),              // td1[31:16]=16 (td1=2^20), tile0
               (int)tile1,                              // tile_dim1 (tile_dim2=0)
               (int)ld,                                 // tensor_dim0_stride[31:0]
               0, 0 };
  i32x4 z4 = {0, 0, 0, 0};
#if defined(__clang_major__) && (__clang_major__ >= 23)
  i32x8 z8 = {0, 0, 0, 0, 0, 0, 0, 0};
  __builtin_amdgcn_tensor_load_to_lds(g0, g1, z4, z4, z8, 0);
#else
  __builtin_amdgcn_tensor_load_to_lds(g0, g1, z4, z4, 0);
#endif
}

// ------------------------------- pack kernels ------------------------------

// Pre-swizzle a B matrix (K x N, bf16) into fragment-native tiles.
__global__ void k_pack_b(const float* __restrict__ src, unsigned short* __restrict__ dst,
                         int K, int N, int mode) {
  int d = blockIdx.x * 256 + threadIdx.x;
  if (d >= K * N) return;
  int tile = d >> 9, r = d & 511, lane = r >> 4, e = r & 15;
  int NT = N >> 4;
  int nt = tile % NT, kt = tile / NT;
  int k = kt * 32 + ((lane >> 4) << 3) + (e & 7) + ((e >> 3) << 4);
  int n = nt * 16 + (lane & 15);
  float val = 0.f;
  switch (mode) {
    case 0: val = src[(size_t)n * 384 + k]; break;                     // vq0^T (384x2048)
    case 1: val = src[(size_t)k * 384 + n]; break;                     // vq0   (2048x384)
    case 2: { int t9 = k / 384, ci = k % 384;                          // w1 (3456x2048)
              val = src[((size_t)n * 384 + ci) * 9 + t9]; } break;
    case 3: val = src[(size_t)n * 2048 + k]; break;                    // w2^T (2048x2048)
    case 4: val = (n < 27) ? src[(size_t)n * 2048 + k] : 0.f; break;   // vq1^T pad->32
    case 5: val = (k < 27) ? src[(size_t)k * 2048 + n] : 0.f; break;   // vq1   pad->32
  }
  dst[d] = f2bf(val);
}

__global__ void k_norms(const float* __restrict__ code, float* __restrict__ norms,
                        int C) {
  __shared__ float red[128];
  int n = blockIdx.x, t = threadIdx.x;
  float s = 0.f;
  for (int c = t; c < C; c += 128) { float v = code[(size_t)n * C + c]; s += v * v; }
  red[t] = s; __syncthreads();
  for (int o = 64; o > 0; o >>= 1) { if (t < o) red[t] += red[t + o]; __syncthreads(); }
  if (t == 0) norms[n] = red[0];
}

// x0: NCHW f32 -> NHWC bf16 ([pixel][384] row-major)
__global__ void k_pack_x0(const float* __restrict__ x0, unsigned short* __restrict__ x0h) {
  int t = blockIdx.x * 256 + threadIdx.x;
  if (t >= 25600 * 384) return;
  int c = t % 384, pix = t / 384;
  int b = pix / 1600, hw = pix % 1600;
  x0h[t] = f2bf(x0[((size_t)b * 384 + c) * 1600 + hw]);
}

// ------------------------- VQ0: scores + softmax ---------------------------
// One WG = 16 rows x 2048 codes. 8 waves, 16 N-tiles each, scores in registers.
__launch_bounds__(256)
__global__ void k_vq0_scores(const unsigned short* __restrict__ x0h,
                             const unsigned short* __restrict__ vq0T,
                             const float* __restrict__ norm0,
                             float* __restrict__ a0,             // assignment0 (NCHW)
                             unsigned short* __restrict__ prob)  // bf16 [pix][2048]
{
  __shared__ __align__(16) unsigned short As[16 * 384];
  __shared__ float redmax[16 * 8];
  __shared__ float redsum[16 * 8];
  int tid = threadIdx.x, lane = tid & 31, wave = tid >> 5;
  int row0 = blockIdx.x * 16;

  for (int i = tid * 8; i < 16 * 384; i += 256 * 8)
    *(u16x8*)(As + i) = *(const u16x8*)(x0h + (size_t)row0 * 384 + i);
  __syncthreads();

  v8f zero = {0, 0, 0, 0, 0, 0, 0, 0};
  v8f acc[16];
#pragma unroll
  for (int j = 0; j < 16; ++j) acc[j] = zero;

  for (int kt = 0; kt < 12; ++kt) {
    v16bf a = a_frag(As, 384, kt * 32, lane);
#pragma unroll
    for (int j = 0; j < 16; ++j) {
      size_t tile = (size_t)kt * 128 + (wave * 16 + j);
      acc[j] = wmma_bf16(a, b_frag(vq0T, tile, lane), acc[j]);
    }
  }

  // scores -> logits:  200*s - 100*||c||^2   (TEMP = 0.01)
#pragma unroll
  for (int j = 0; j < 16; ++j) {
    float nrm = norm0[(wave * 16 + j) * 16 + (lane & 15)];
#pragma unroll
    for (int v = 0; v < 8; ++v) acc[j][v] = 200.0f * acc[j][v] - 100.0f * nrm;
  }

  int half = lane >> 4;
  float rmax[8], rsum[8];
#pragma unroll
  for (int v = 0; v < 8; ++v) {
    float mx = -1e30f;
#pragma unroll
    for (int j = 0; j < 16; ++j) mx = fmaxf(mx, acc[j][v]);
    for (int m = 1; m <= 8; m <<= 1) mx = fmaxf(mx, __shfl_xor(mx, m, 32));
    if ((lane & 15) == 0) redmax[(v + 8 * half) * 8 + wave] = mx;
  }
  __syncthreads();
#pragma unroll
  for (int v = 0; v < 8; ++v) {
    float mx = -1e30f;
    for (int w = 0; w < 8; ++w) mx = fmaxf(mx, redmax[(v + 8 * half) * 8 + w]);
    rmax[v] = mx;
    rsum[v] = 0.f;
  }
#pragma unroll
  for (int j = 0; j < 16; ++j)
#pragma unroll
    for (int v = 0; v < 8; ++v) {
      float e = __expf(acc[j][v] - rmax[v]);
      acc[j][v] = e;
      rsum[v] += e;
    }
#pragma unroll
  for (int v = 0; v < 8; ++v) {
    float s = rsum[v];
    for (int m = 1; m <= 8; m <<= 1) s += __shfl_xor(s, m, 32);
    if ((lane & 15) == 0) redsum[(v + 8 * half) * 8 + wave] = s;
  }
  __syncthreads();
#pragma unroll
  for (int v = 0; v < 8; ++v) {
    float s = 0.f;
    for (int w = 0; w < 8; ++w) s += redsum[(v + 8 * half) * 8 + w];
    rsum[v] = 1.0f / s;
  }
#pragma unroll
  for (int j = 0; j < 16; ++j) {
    int n = (wave * 16 + j) * 16 + (lane & 15);
#pragma unroll
    for (int v = 0; v < 8; ++v) {
      float p = acc[j][v] * rsum[v];
      int pix = row0 + v + 8 * half;
      int b = pix / 1600, hw = pix % 1600;
      a0[((size_t)b * 2048 + n) * 1600 + hw] = p;
      prob[(size_t)pix * 2048 + n] = f2bf(p);
    }
  }
}

// --------------------- VQ0 apply: q = prob @ code --------------------------
// A tiles (16x256 bf16) staged by the Tensor Data Mover, double-buffered.
__launch_bounds__(256)
__global__ void k_vq0_apply(const unsigned short* __restrict__ prob,
                            const unsigned short* __restrict__ vq0p,
                            float* __restrict__ qx0,              // NCHW f32
                            unsigned short* __restrict__ qx0h)    // NHWC bf16
{
  __shared__ __align__(16) unsigned short As[2][16 * 256];
  int tid = threadIdx.x, lane = tid & 31, wave = tid >> 5;
  int row0 = blockIdx.x * 16;
  const unsigned short* gbase = prob + (size_t)row0 * 2048;
  v8f zero = {0, 0, 0, 0, 0, 0, 0, 0};
  v8f acc[3];
#pragma unroll
  for (int j = 0; j < 3; ++j) acc[j] = zero;

  if (wave == 0)
    tdm_load_tile_bf16((unsigned)(uintptr_t)(void*)As[0], gbase, 256, 16, 2048);

  for (int cc = 0; cc < 8; ++cc) {
    if (wave == 0) {
      if (cc < 7) {
        tdm_load_tile_bf16((unsigned)(uintptr_t)(void*)As[(cc + 1) & 1],
                           gbase + (cc + 1) * 256, 256, 16, 2048);
        __builtin_amdgcn_s_wait_tensorcnt(1);   // chunk cc is resident
      } else {
        __builtin_amdgcn_s_wait_tensorcnt(0);
      }
    }
    __syncthreads();
    const unsigned short* Ab = As[cc & 1];
    for (int kt = 0; kt < 8; ++kt) {
      v16bf a = a_frag(Ab, 256, kt * 32, lane);
#pragma unroll
      for (int j = 0; j < 3; ++j) {
        size_t tile = (size_t)(cc * 8 + kt) * 24 + (wave * 3 + j);
        acc[j] = wmma_bf16(a, b_frag(vq0p, tile, lane), acc[j]);
      }
    }
    __syncthreads();   // readers done before this buffer is re-filled
  }
  int half = lane >> 4;
#pragma unroll
  for (int j = 0; j < 3; ++j) {
    int n = (wave * 3 + j) * 16 + (lane & 15);
#pragma unroll
    for (int v = 0; v < 8; ++v) {
      float q = acc[j][v];
      int pix = row0 + v + 8 * half;
      int b = pix / 1600, hw = pix % 1600;
      qx0[((size_t)b * 384 + n) * 1600 + hw] = q;
      qx0h[(size_t)pix * 384 + n] = f2bf(q);
    }
  }
}

// ------------- conv3x3 (reflect pad) + bias + LeakyReLU, implicit GEMM -----
__launch_bounds__(256)
__global__ void k_conv3x3(const unsigned short* __restrict__ qx0h,
                          const unsigned short* __restrict__ w1p,
                          const float* __restrict__ b1,
                          unsigned short* __restrict__ hbuf)   // bf16 [pix][2048]
{
  __shared__ __align__(16) unsigned short As[16 * 384];
  int tid = threadIdx.x, lane = tid & 31, wave = tid >> 5;
  int row0 = blockIdx.x * 16;
  v8f zero = {0, 0, 0, 0, 0, 0, 0, 0};
  v8f acc[16];
#pragma unroll
  for (int j = 0; j < 16; ++j) acc[j] = zero;

  for (int t9 = 0; t9 < 9; ++t9) {
    int ky = t9 / 3 - 1, kx = t9 % 3 - 1;
    __syncthreads();
    for (int i = tid * 8; i < 16 * 384; i += 256 * 8) {
      int r = i / 384, c = i % 384;
      int pix = row0 + r, b = pix / 1600, rem = pix % 1600;
      int y = refl(rem / 40 + ky), x = refl(rem % 40 + kx);
      *(u16x8*)(As + i) =
          *(const u16x8*)(qx0h + ((size_t)((b * 40 + y) * 40 + x)) * 384 + c);
    }
    __syncthreads();
    for (int kt = 0; kt < 12; ++kt) {
      v16bf a = a_frag(As, 384, kt * 32, lane);
      __builtin_prefetch(w1p + ((size_t)(t9 * 12 + kt + 1) * 128 + wave * 16) * 512, 0, 1);
#pragma unroll
      for (int j = 0; j < 16; ++j) {
        size_t tile = (size_t)(t9 * 12 + kt) * 128 + (wave * 16 + j);
        acc[j] = wmma_bf16(a, b_frag(w1p, tile, lane), acc[j]);
      }
    }
  }
  int half = lane >> 4;
#pragma unroll
  for (int j = 0; j < 16; ++j) {
    int n = (wave * 16 + j) * 16 + (lane & 15);
    float bias = b1[n];
#pragma unroll
    for (int v = 0; v < 8; ++v) {
      float h = acc[j][v] + bias;
      h = h >= 0.f ? h : 0.1f * h;                  // LeakyReLU(0.1)
      int pix = row0 + v + 8 * half;
      hbuf[(size_t)pix * 2048 + n] = f2bf(h);
    }
  }
}

// ----------------------- conv1x1 + bias -> x3 ------------------------------
// A tiles staged by TDM, double-buffered; B streamed with prefetch.
__launch_bounds__(256)
__global__ void k_conv1x1(const unsigned short* __restrict__ hbuf,
                          const unsigned short* __restrict__ w2p,
                          const float* __restrict__ b2,
                          float* __restrict__ x3,               // NCHW f32 (output)
                          unsigned short* __restrict__ x3h)     // bf16 [pix][2048]
{
  __shared__ __align__(16) unsigned short As[2][16 * 256];
  int tid = threadIdx.x, lane = tid & 31, wave = tid >> 5;
  int row0 = blockIdx.x * 16;
  const unsigned short* gbase = hbuf + (size_t)row0 * 2048;
  v8f zero = {0, 0, 0, 0, 0, 0, 0, 0};
  v8f acc[16];
#pragma unroll
  for (int j = 0; j < 16; ++j) acc[j] = zero;

  if (wave == 0)
    tdm_load_tile_bf16((unsigned)(uintptr_t)(void*)As[0], gbase, 256, 16, 2048);

  for (int cc = 0; cc < 8; ++cc) {
    if (wave == 0) {
      if (cc < 7) {
        tdm_load_tile_bf16((unsigned)(uintptr_t)(void*)As[(cc + 1) & 1],
                           gbase + (cc + 1) * 256, 256, 16, 2048);
        __builtin_amdgcn_s_wait_tensorcnt(1);
      } else {
        __builtin_amdgcn_s_wait_tensorcnt(0);
      }
    }
    __syncthreads();
    const unsigned short* Ab = As[cc & 1];
    for (int kt = 0; kt < 8; ++kt) {
      v16bf a = a_frag(Ab, 256, kt * 32, lane);
      __builtin_prefetch(w2p + ((size_t)(cc * 8 + kt + 1) * 128 + wave * 16) * 512, 0, 1);
#pragma unroll
      for (int j = 0; j < 16; ++j) {
        size_t tile = (size_t)(cc * 8 + kt) * 128 + (wave * 16 + j);
        acc[j] = wmma_bf16(a, b_frag(w2p, tile, lane), acc[j]);
      }
    }
    __syncthreads();
  }
  int half = lane >> 4;
#pragma unroll
  for (int j = 0; j < 16; ++j) {
    int n = (wave * 16 + j) * 16 + (lane & 15);
    float bias = b2[n];
#pragma unroll
    for (int v = 0; v < 8; ++v) {
      float val = acc[j][v] + bias;
      int pix = row0 + v + 8 * half;
      int b = pix / 1600, hw = pix % 1600;
      x3[((size_t)b * 2048 + n) * 1600 + hw] = val;
      x3h[(size_t)pix * 2048 + n] = f2bf(val);
    }
  }
}

// ---------------- VQ1 fused: dist GEMM + softmax + apply GEMM --------------
// K = 27 codes padded to 32 (padded logits forced to -inf).
__launch_bounds__(256)
__global__ void k_vq1(const unsigned short* __restrict__ x3h,
                      const unsigned short* __restrict__ vq1p1,  // B: [c][code32]
                      const unsigned short* __restrict__ vq1p2,  // B: [code32][c]
                      const float* __restrict__ norm1,
                      float* __restrict__ a3,                    // (b,27,h,w)
                      float* __restrict__ qx3)                   // NCHW f32
{
  __shared__ float scpart[8 * 512];
  __shared__ float lg[512];
  __shared__ __align__(16) unsigned short pb[16 * 32];
  int tid = threadIdx.x, lane = tid & 31, wave = tid >> 5;
  int row0 = blockIdx.x * 16;
  v8f zero = {0, 0, 0, 0, 0, 0, 0, 0};

  // Phase 1: each wave accumulates a 256-wide slice of the inner dim.
  v8f acc0 = zero, acc1 = zero;
  for (int kt = 0; kt < 8; ++kt) {
    int k0 = wave * 256 + kt * 32;
    int m = lane & 15, kh = (lane >> 4) << 3;
    const unsigned short* p = x3h + (size_t)(row0 + m) * 2048 + k0 + kh;
    V16 av;
    av.h[0] = *(const u16x8*)(p);
    av.h[1] = *(const u16x8*)(p + 16);
    size_t tb = (size_t)(wave * 8 + kt) * 2;
    acc0 = wmma_bf16(av.v, b_frag(vq1p1, tb + 0, lane), acc0);
    acc1 = wmma_bf16(av.v, b_frag(vq1p1, tb + 1, lane), acc1);
  }
#pragma unroll
  for (int v = 0; v < 8; ++v) {
    scpart[wave * 512 + 0 * 256 + v * 32 + lane] = acc0[v];
    scpart[wave * 512 + 1 * 256 + v * 32 + lane] = acc1[v];
  }
  __syncthreads();

  // Phase 2: cross-wave reduce + logits (pad codes masked to -inf)
  for (int idx = tid; idx < 512; idx += 256) {
    float s = 0.f;
#pragma unroll
    for (int w = 0; w < 8; ++w) s += scpart[w * 512 + idx];
    int j = idx >> 8, rem = idx & 255, ln = rem & 31;
    int n = j * 16 + (ln & 15);
    lg[idx] = (n < 27) ? (200.0f * s - 100.0f * norm1[n]) : -1e30f;
  }
  __syncthreads();

  // Softmax over 32 per row; write assignment3 + bf16 prob tile to LDS.
  if (tid < 16) {
    int m = tid, v = m & 7, half = m >> 3;
    float vals[32], mx = -1e30f;
#pragma unroll
    for (int n = 0; n < 32; ++n) {
      float s = lg[(n >> 4) * 256 + v * 32 + half * 16 + (n & 15)];
      vals[n] = s; mx = fmaxf(mx, s);
    }
    float sum = 0.f;
#pragma unroll
    for (int n = 0; n < 32; ++n) { float e = __expf(vals[n] - mx); vals[n] = e; sum += e; }
    float inv = 1.0f / sum;
    int pix = row0 + m, b = pix / 1600, hw = pix % 1600;
#pragma unroll
    for (int n = 0; n < 32; ++n) {
      float pv = vals[n] * inv;
      pb[m * 32 + n] = f2bf(pv);
      if (n < 27) a3[((size_t)b * 27 + n) * 1600 + hw] = pv;
    }
  }
  __syncthreads();

  // Phase 3: qx3 = P(16x32) @ vq1(32x2048); single k-step.
  v8f accq[16];
#pragma unroll
  for (int j = 0; j < 16; ++j) accq[j] = zero;
  v16bf a = a_frag(pb, 32, 0, lane);
#pragma unroll
  for (int j = 0; j < 16; ++j)
    accq[j] = wmma_bf16(a, b_frag(vq1p2, (size_t)(wave * 16 + j), lane), accq[j]);

  int half = lane >> 4;
#pragma unroll
  for (int j = 0; j < 16; ++j) {
    int n = (wave * 16 + j) * 16 + (lane & 15);
#pragma unroll
    for (int v = 0; v < 8; ++v) {
      int pix = row0 + v + 8 * half;
      int b = pix / 1600, hw = pix % 1600;
      qx3[((size_t)b * 2048 + n) * 1600 + hw] = accq[j][v];
    }
  }
}

// ------------------------------- launcher ----------------------------------

extern "C" void kernel_launch(void* const* d_in, const int* in_sizes, int n_in,
                              void* d_out, int out_size, void* d_ws, size_t ws_size,
                              hipStream_t stream) {
  (void)in_sizes; (void)n_in; (void)out_size; (void)ws_size;
  const float* x0  = (const float*)d_in[0];
  const float* vq0 = (const float*)d_in[1];
  const float* vq1 = (const float*)d_in[2];
  const float* w1  = (const float*)d_in[3];
  const float* b1  = (const float*)d_in[4];
  const float* w2  = (const float*)d_in[5];
  const float* b2  = (const float*)d_in[6];
  float* out = (float*)d_out;

  // Output tuple offsets (floats): x0, x3, qx0, qx3, assignment0, assignment3
  const size_t O_X3 = 9830400, O_QX0 = 62259200, O_QX3 = 72089600,
               O_A0 = 124518400, O_A3 = 176947200;

  char* w = (char*)d_ws;
  auto alloc = [&](size_t bytes) {
    char* p = w; w += (bytes + 255) & ~(size_t)255; return p;
  };
  unsigned short* x0h   = (unsigned short*)alloc(25600ull * 384 * 2);
  unsigned short* qx0h  = (unsigned short*)alloc(25600ull * 384 * 2);
  unsigned short* prob  = (unsigned short*)alloc(25600ull * 2048 * 2);
  unsigned short* hbuf  = (unsigned short*)alloc(25600ull * 2048 * 2);
  unsigned short* x3h   = (unsigned short*)alloc(25600ull * 2048 * 2);
  unsigned short* vq0T  = (unsigned short*)alloc(384ull * 2048 * 2);
  unsigned short* vq0p  = (unsigned short*)alloc(2048ull * 384 * 2);
  unsigned short* w1p   = (unsigned short*)alloc(3456ull * 2048 * 2);
  unsigned short* w2p   = (unsigned short*)alloc(2048ull * 2048 * 2);
  unsigned short* vq1p1 = (unsigned short*)alloc(2048ull * 32 * 2);
  unsigned short* vq1p2 = (unsigned short*)alloc(32ull * 2048 * 2);
  float* norm0 = (float*)alloc(2048 * 4);
  float* norm1 = (float*)alloc(32 * 4);

  hipMemcpyAsync(out, x0, 9830400ull * 4, hipMemcpyDeviceToDevice, stream);

  auto nb = [](long long n) { return (unsigned)((n + 255) / 256); };
  k_pack_x0<<<nb(25600ll * 384), 256, 0, stream>>>(x0, x0h);
  k_pack_b<<<nb(384ll * 2048), 256, 0, stream>>>(vq0, vq0T, 384, 2048, 0);
  k_pack_b<<<nb(2048ll * 384), 256, 0, stream>>>(vq0, vq0p, 2048, 384, 1);
  k_pack_b<<<nb(3456ll * 2048), 256, 0, stream>>>(w1, w1p, 3456, 2048, 2);
  k_pack_b<<<nb(2048ll * 2048), 256, 0, stream>>>(w2, w2p, 2048, 2048, 3);
  k_pack_b<<<nb(2048ll * 32), 256, 0, stream>>>(vq1, vq1p1, 2048, 32, 4);
  k_pack_b<<<nb(32ll * 2048), 256, 0, stream>>>(vq1, vq1p2, 32, 2048, 5);
  k_norms<<<2048, 128, 0, stream>>>(vq0, norm0, 384);
  k_norms<<<27, 128, 0, stream>>>(vq1, norm1, 2048);

  k_vq0_scores<<<1600, 256, 0, stream>>>(x0h, vq0T, norm0, out + O_A0, prob);
  k_vq0_apply<<<1600, 256, 0, stream>>>(prob, vq0p, out + O_QX0, qx0h);
  k_conv3x3<<<1600, 256, 0, stream>>>(qx0h, w1p, b1, hbuf);
  k_conv1x1<<<1600, 256, 0, stream>>>(hbuf, w2p, b2, out + O_X3, x3h);
  k_vq1<<<1600, 256, 0, stream>>>(x3h, vq1p1, vq1p2, norm1, out + O_A3, out + O_QX3);
}
